// SwinBlock_38482906972358
// MI455X (gfx1250) — matrix operand order repaired and checked by reading, not compiled, not run on hardware
//
#include <hip/hip_runtime.h>
#include <hip/hip_bf16.h>

typedef __attribute__((ext_vector_type(16))) _Float16 v16h;
typedef __attribute__((ext_vector_type(8)))  _Float16 h8;
typedef __attribute__((ext_vector_type(8)))  float    v8f;

#define HH 56
#define SS 3
#define NWIN 512          // total windows = 8 * 64
#define NTOK 25088        // 8 * 56 * 56

// ---------------- WMMA fragment helpers (gfx1250 wave32 layouts) ----------------

__device__ __forceinline__ v16h cat8(h8 lo, h8 hi) {
  return __builtin_shufflevector(lo, hi, 0,1,2,3,4,5,6,7,8,9,10,11,12,13,14,15);
}

// A matrix 16x32 f16, row-major, leading dim ld (in halfs).
// lane l: row = l&15, kb = (l>>4)*16, 16 contiguous k values.
__device__ __forceinline__ v16h load_a16(const _Float16* p0, int ld) {
  int l = threadIdx.x & 31;
  const _Float16* p = p0 + (size_t)(l & 15) * ld + ((l >> 4) << 4);
  h8 lo = *(const h8*)p;
  h8 hi = *(const h8*)(p + 8);
  return cat8(lo, hi);
}

// B matrix 32x16 f16 given as B^T stored [N][K] row-major (leading dim ld = K).
// lane l: col = l&15, kb = (l>>4)*16, 16 contiguous k values.
__device__ __forceinline__ v16h load_b16(const _Float16* p0, int ld) {
  int l = threadIdx.x & 31;
  const _Float16* p = p0 + (size_t)(l & 15) * ld + ((l >> 4) << 4);
  h8 lo = *(const h8*)p;
  h8 hi = *(const h8*)(p + 8);
  return cat8(lo, hi);
}

// B matrix 32x16 f16 stored [K][N] row-major (leading dim ld = N): strided gather.
__device__ __forceinline__ v16h load_b16_kn(const _Float16* p0, int ld) {
  int l = threadIdx.x & 31;
  const _Float16* p = p0 + (size_t)((l >> 4) << 4) * ld + (l & 15);
  v16h b;
#pragma unroll
  for (int e = 0; e < 16; ++e) b[e] = p[(size_t)e * ld];
  return b;
}

__device__ __forceinline__ v8f wmma32(v16h a, v16h b, v8f c) {
  return __builtin_amdgcn_wmma_f32_16x16x32_f16(false, a, false, b, (short)0, c, false, false);
}

// Double-buffered wave-strip GEMM core: one wave computes a 32(M) x 64(N) strip.
// A: [M][K] f16 row-major; Bt: [N][K] f16 (pre-transposed weights); f32 accum.
// K is compile-time so the loop fully unrolls and next-step fragment loads are
// issued before the current step's WMMAs (overlaps global latency with matrix pipe).
template <int K>
__device__ __forceinline__ void gemm_core2(const _Float16* __restrict__ A,
                                           const _Float16* __restrict__ Bt,
                                           size_t row0, size_t col0, v8f acc[2][4]) {
  v16h a0 = load_a16(A + row0 * K, K);
  v16h a1 = load_a16(A + (row0 + 16) * K, K);
  v16h b[4];
#pragma unroll
  for (int j = 0; j < 4; ++j) b[j] = load_b16(Bt + (col0 + 16 * j) * K, K);
#pragma unroll
  for (int kk = 0; kk < K; kk += 32) {
    v16h na0, na1, nb[4];
    if (kk + 32 < K) {  // prefetch next K-step fragments
      na0 = load_a16(A + row0 * K + kk + 32, K);
      na1 = load_a16(A + (row0 + 16) * K + kk + 32, K);
#pragma unroll
      for (int j = 0; j < 4; ++j) nb[j] = load_b16(Bt + (col0 + 16 * j) * K + kk + 32, K);
    }
#pragma unroll
    for (int j = 0; j < 4; ++j) {
      acc[0][j] = wmma32(a0, b[j], acc[0][j]);
      acc[1][j] = wmma32(a1, b[j], acc[1][j]);
    }
    if (kk + 32 < K) {
      a0 = na0; a1 = na1;
#pragma unroll
      for (int j = 0; j < 4; ++j) b[j] = nb[j];
    }
  }
}

__device__ __forceinline__ float wave_sum(float v) {
#pragma unroll
  for (int off = 16; off > 0; off >>= 1) v += __shfl_xor(v, off, 32);
  return v;
}

__device__ __forceinline__ float gelu_tanh(float x) {
  float x3 = x * x * x;
  return 0.5f * x * (1.0f + tanhf(0.7978845608028654f * (x + 0.044715f * x3)));
}

// ---------------- weight convert+transpose: out[n*K + k] = (f16) w[k*N + n] ----------------
__global__ void wcvt_kernel(const float* __restrict__ w, _Float16* __restrict__ o, int K, int N) {
  size_t i = (size_t)blockIdx.x * 256 + threadIdx.x;
  if (i >= (size_t)K * N) return;
  int n = (int)(i / K), kk = (int)(i % K);
  o[i] = (_Float16)w[(size_t)kk * N + n];
}

// ---------------- LN1 + roll(-3,-3) + window partition -> winX [512][64][512] f16 ----------------
__global__ void ln1_win_kernel(const float* __restrict__ x, const float* __restrict__ g,
                               const float* __restrict__ bt, _Float16* __restrict__ winX) {
  int wave = threadIdx.x >> 5, lane = threadIdx.x & 31;
  int t = blockIdx.x * 8 + wave;            // 0..25087
  int win = t / 49, tok = t - win * 49;
  int b = win >> 6, widx = win & 63;
  int ty = tok / 7, tx = tok - ty * 7;
  int sy = (widx >> 3) * 7 + ty + SS; if (sy >= HH) sy -= HH;
  int sx = (widx & 7) * 7 + tx + SS;  if (sx >= HH) sx -= HH;
  const float* src = x + (((size_t)b * 3136) + sy * 56 + sx) * 512;
  float vals[16]; float s = 0.f;
#pragma unroll
  for (int i = 0; i < 16; ++i) { vals[i] = src[lane + 32 * i]; s += vals[i]; }
  s = wave_sum(s);
  float mu = s * (1.0f / 512.0f);
  float vs = 0.f;
#pragma unroll
  for (int i = 0; i < 16; ++i) { float d = vals[i] - mu; vs += d * d; }
  vs = wave_sum(vs);
  float rs = rsqrtf(vs * (1.0f / 512.0f) + 1e-5f);
  _Float16* dst = winX + ((size_t)win * 64 + tok) * 512;
#pragma unroll
  for (int i = 0; i < 16; ++i) {
    int c = lane + 32 * i;
    dst[c] = (_Float16)((vals[i] - mu) * rs * g[c] + bt[c]);
  }
}

// ---------------- LN2 over x2 -> z16 f16 ----------------
__global__ void ln2_kernel(const float* __restrict__ x2, const float* __restrict__ g,
                           const float* __restrict__ bt, _Float16* __restrict__ z16) {
  int wave = threadIdx.x >> 5, lane = threadIdx.x & 31;
  int t = blockIdx.x * 8 + wave;
  const float* src = x2 + (size_t)t * 512;
  float vals[16]; float s = 0.f;
#pragma unroll
  for (int i = 0; i < 16; ++i) { vals[i] = src[lane + 32 * i]; s += vals[i]; }
  s = wave_sum(s);
  float mu = s * (1.0f / 512.0f);
  float vs = 0.f;
#pragma unroll
  for (int i = 0; i < 16; ++i) { float d = vals[i] - mu; vs += d * d; }
  vs = wave_sum(vs);
  float rs = rsqrtf(vs * (1.0f / 512.0f) + 1e-5f);
  _Float16* dst = z16 + (size_t)t * 512;
#pragma unroll
  for (int i = 0; i < 16; ++i) {
    int c = lane + 32 * i;
    dst[c] = (_Float16)((vals[i] - mu) * rs * g[c] + bt[c]);
  }
}

// ---------------- QKV GEMM: winX[32768x512] @ qkv_w[512x1536] -> q/k/v [win][head][64][32] ----------------
__global__ void gemm_qkv(const _Float16* __restrict__ A, const _Float16* __restrict__ Bt,
                         const float* __restrict__ bias,
                         _Float16* __restrict__ q, _Float16* __restrict__ k, _Float16* __restrict__ v) {
  int wave = threadIdx.x >> 5, lane = threadIdx.x & 31;
  size_t row0 = (size_t)blockIdx.y * 128 + (size_t)(wave & 3) * 32;
  size_t col0 = (size_t)blockIdx.x * 128 + (size_t)(wave >> 2) * 64;
  v8f acc[2][4] = {};
  gemm_core2<512>(A, Bt, row0, col0, acc);
  int nn = lane & 15, mb = (lane >> 4) * 8;
#pragma unroll
  for (int mi = 0; mi < 2; ++mi)
#pragma unroll
    for (int j = 0; j < 4; ++j)
#pragma unroll
      for (int r = 0; r < 8; ++r) {
        int n = (int)col0 + j * 16 + nn;
        int m = (int)row0 + mi * 16 + mb + r;
        float cv = acc[mi][j][r] + bias[n];
        int part = n >> 9, rem = n & 511;
        int head = rem >> 5, dim = rem & 31;
        int win = m >> 6, tok = m & 63;
        _Float16* dst = (part == 0) ? q : (part == 1 ? k : v);
        if (part == 0) cv *= 0.17677669529663687f;   // 1/sqrt(32)
        dst[(((size_t)win * 16 + head) * 64 + tok) * 32 + dim] = (_Float16)cv;
      }
}

// ---------------- fused window attention: one block per (window, head) ----------------
__global__ void attn_kernel(const _Float16* __restrict__ q, const _Float16* __restrict__ k,
                            const _Float16* __restrict__ v, _Float16* __restrict__ ao) {
  __shared__ float    sS[64][64];
  __shared__ _Float16 sP[64][64];
  __shared__ int      rid[49];
  int wh = blockIdx.x;
  int win = wh >> 4, head = wh & 15;
  int wave = threadIdx.x >> 5, lane = threadIdx.x & 31;
  const _Float16* qb = q + (size_t)wh * (64 * 32);
  const _Float16* kb = k + (size_t)wh * (64 * 32);
  const _Float16* vb = v + (size_t)wh * (64 * 32);

  if (threadIdx.x < 49) {  // shifted-window region ids (mask built on unrolled grid)
    int tok = threadIdx.x;
    int widx = win & 63;
    int ty = tok / 7, tx = tok - ty * 7;
    int r = (widx >> 3) * 7 + ty, c = (widx & 7) * 7 + tx;
    int rr = r < 49 ? 0 : (r < 53 ? 1 : 2);
    int rc = c < 49 ? 0 : (c < 53 ? 1 : 2);
    rid[tok] = rr * 3 + rc;
  }

  // scores: S = (q*scale) @ k^T  -- wave owns row-tile `wave`, 4 col tiles
  v16h a = load_a16(qb + wave * 16 * 32, 32);
  v8f acc[4];
#pragma unroll
  for (int j = 0; j < 4; ++j) {
    v8f z = {};
    v16h bf = load_b16(kb + j * 16 * 32, 32);  // k rows contiguous == (k^T) cols
    acc[j] = wmma32(a, bf, z);
  }
  int nn = lane & 15, mb = (lane >> 4) * 8;
#pragma unroll
  for (int j = 0; j < 4; ++j)
#pragma unroll
    for (int r = 0; r < 8; ++r)
      sS[wave * 16 + mb + r][j * 16 + nn] = acc[j][r];
  __syncthreads();

  // masked softmax, one thread per row
  int t = threadIdx.x;
  if (t < 64) {
    if (t < 49) {
      int myr = rid[t];
      float mx = -3.0e38f;
      for (int j = 0; j < 49; ++j) {
        float val = (rid[j] == myr) ? sS[t][j] : -3.0e38f;
        mx = fmaxf(mx, val);
      }
      float sum = 0.f;
      for (int j = 0; j < 49; ++j) {
        float e = (rid[j] == myr) ? __expf(sS[t][j] - mx) : 0.0f;
        sS[t][j] = e; sum += e;
      }
      float inv = 1.0f / sum;
      for (int j = 0; j < 49; ++j) sP[t][j] = (_Float16)(sS[t][j] * inv);
      for (int j = 49; j < 64; ++j) sP[t][j] = (_Float16)0.0f;
    } else {
      for (int j = 0; j < 64; ++j) sP[t][j] = (_Float16)0.0f;
    }
  }
  __syncthreads();

  // out = P @ V : wave owns 16 rows x 32 cols, K = 64 (2 steps)
  v8f o[2] = {};
#pragma unroll
  for (int kk = 0; kk < 64; kk += 32) {
    const _Float16* pa = &sP[wave * 16 + (lane & 15)][kk + ((lane >> 4) << 4)];
    h8 lo = *(const h8*)pa;
    h8 hi = *(const h8*)(pa + 8);
    v16h af = cat8(lo, hi);
#pragma unroll
    for (int j = 0; j < 2; ++j) {
      v16h bf = load_b16_kn(vb + (size_t)kk * 32 + j * 16, 32);
      o[j] = wmma32(af, bf, o[j]);
    }
  }
#pragma unroll
  for (int j = 0; j < 2; ++j)
#pragma unroll
    for (int r = 0; r < 8; ++r) {
      int tokr = wave * 16 + mb + r;
      int dim = j * 16 + nn;
      ao[((size_t)win * 64 + tokr) * 512 + head * 32 + dim] = (_Float16)o[j][r];
    }
}

// ---------------- proj GEMM + window reverse + roll(+3,+3) + residual -> x2 f32 ----------------
__global__ void gemm_proj(const _Float16* __restrict__ A, const _Float16* __restrict__ Bt,
                          const float* __restrict__ bias, const float* __restrict__ xin,
                          float* __restrict__ x2) {
  int wave = threadIdx.x >> 5, lane = threadIdx.x & 31;
  size_t row0 = (size_t)blockIdx.y * 128 + (size_t)(wave & 3) * 32;
  size_t col0 = (size_t)blockIdx.x * 128 + (size_t)(wave >> 2) * 64;
  v8f acc[2][4] = {};
  gemm_core2<512>(A, Bt, row0, col0, acc);
  int nn = lane & 15, mb = (lane >> 4) * 8;
#pragma unroll
  for (int mi = 0; mi < 2; ++mi)
#pragma unroll
    for (int j = 0; j < 4; ++j)
#pragma unroll
      for (int r = 0; r < 8; ++r) {
        int n = (int)col0 + j * 16 + nn;
        int m = (int)row0 + mi * 16 + mb + r;
        int win = m >> 6, tok = m & 63;
        if (tok < 49) {
          int b = win >> 6, widx = win & 63;
          int ty = tok / 7, tx = tok - ty * 7;
          int sy = (widx >> 3) * 7 + ty + SS; if (sy >= HH) sy -= HH;
          int sx = (widx & 7) * 7 + tx + SS;  if (sx >= HH) sx -= HH;
          size_t idx = (((size_t)b * 3136) + sy * 56 + sx) * 512 + n;
          x2[idx] = xin[idx] + acc[mi][j][r] + bias[n];
        }
      }
}

// ---------------- FC1 GEMM + GELU -> h16 f16 ----------------
__global__ void gemm_fc1(const _Float16* __restrict__ A, const _Float16* __restrict__ Bt,
                         const float* __restrict__ bias, _Float16* __restrict__ h16) {
  int wave = threadIdx.x >> 5, lane = threadIdx.x & 31;
  size_t row0 = (size_t)blockIdx.y * 128 + (size_t)(wave & 3) * 32;
  size_t col0 = (size_t)blockIdx.x * 128 + (size_t)(wave >> 2) * 64;
  v8f acc[2][4] = {};
  gemm_core2<512>(A, Bt, row0, col0, acc);
  int nn = lane & 15, mb = (lane >> 4) * 8;
#pragma unroll
  for (int mi = 0; mi < 2; ++mi)
#pragma unroll
    for (int j = 0; j < 4; ++j)
#pragma unroll
      for (int r = 0; r < 8; ++r) {
        int n = (int)col0 + j * 16 + nn;
        size_t m = row0 + mi * 16 + mb + r;
        h16[m * 1024 + n] = (_Float16)gelu_tanh(acc[mi][j][r] + bias[n]);
      }
}

// ---------------- FC2 GEMM + bias + residual(x2) -> out f32 ----------------
__global__ void gemm_fc2(const _Float16* __restrict__ A, const _Float16* __restrict__ Bt,
                         const float* __restrict__ bias, const float* __restrict__ x2,
                         float* __restrict__ out) {
  int wave = threadIdx.x >> 5, lane = threadIdx.x & 31;
  size_t row0 = (size_t)blockIdx.y * 128 + (size_t)(wave & 3) * 32;
  size_t col0 = (size_t)blockIdx.x * 128 + (size_t)(wave >> 2) * 64;
  v8f acc[2][4] = {};
  gemm_core2<1024>(A, Bt, row0, col0, acc);
  int nn = lane & 15, mb = (lane >> 4) * 8;
#pragma unroll
  for (int mi = 0; mi < 2; ++mi)
#pragma unroll
    for (int j = 0; j < 4; ++j)
#pragma unroll
      for (int r = 0; r < 8; ++r) {
        int n = (int)col0 + j * 16 + nn;
        size_t m = row0 + mi * 16 + mb + r;
        size_t idx = m * 512 + n;
        out[idx] = x2[idx] + acc[mi][j][r] + bias[n];
      }
}

extern "C" void kernel_launch(void* const* d_in, const int* in_sizes, int n_in,
                              void* d_out, int out_size, void* d_ws, size_t ws_size,
                              hipStream_t stream) {
  (void)in_sizes; (void)n_in; (void)out_size; (void)ws_size;
  const float* x     = (const float*)d_in[0];
  const float* ln1w  = (const float*)d_in[1];
  const float* ln1b  = (const float*)d_in[2];
  const float* qkvw  = (const float*)d_in[3];
  const float* qkvb  = (const float*)d_in[4];
  const float* projw = (const float*)d_in[5];
  const float* projb = (const float*)d_in[6];
  const float* ln2w  = (const float*)d_in[7];
  const float* ln2b  = (const float*)d_in[8];
  const float* fc1w  = (const float*)d_in[9];
  const float* fc1b  = (const float*)d_in[10];
  const float* fc2w  = (const float*)d_in[11];
  const float* fc2b  = (const float*)d_in[12];
  float* out = (float*)d_out;

  char* ws = (char*)d_ws;
  size_t off = 0;
  auto alloc = [&](size_t bytes) {
    char* p = ws + off;
    off = (off + bytes + 255) & ~(size_t)255;
    return p;
  };
  const size_t actHalfs = (size_t)NWIN * 64 * 512;  // 16.78M halfs
  _Float16* wq_t  = (_Float16*)alloc((size_t)1536 * 512 * 2);
  _Float16* wp_t  = (_Float16*)alloc((size_t)512 * 512 * 2);
  _Float16* wf1_t = (_Float16*)alloc((size_t)1024 * 512 * 2);
  _Float16* wf2_t = (_Float16*)alloc((size_t)512 * 1024 * 2);
  _Float16* winX  = (_Float16*)alloc(actHalfs * 2);
  _Float16* qb    = (_Float16*)alloc(actHalfs * 2);
  _Float16* kb    = (_Float16*)alloc(actHalfs * 2);
  _Float16* vb    = (_Float16*)alloc(actHalfs * 2);  // kb+vb contiguous: reused as h16
  float*    x2    = (float*)alloc((size_t)NTOK * 512 * 4);
  _Float16* attnO = winX;   // winX dead after QKV GEMM
  _Float16* z16   = qb;     // q dead after attention
  _Float16* h16   = kb;     // k,v dead after attention (25088*1024 halfs fit in kb+vb)
  (void)vb;

  // zero winX so pad token rows (49..63) are clean
  hipMemsetAsync(winX, 0, actHalfs * 2, stream);

  // weight convert + transpose to f16 [N][K]
  wcvt_kernel<<<(1536 * 512 + 255) / 256, 256, 0, stream>>>(qkvw, wq_t, 512, 1536);
  wcvt_kernel<<<(512 * 512 + 255) / 256, 256, 0, stream>>>(projw, wp_t, 512, 512);
  wcvt_kernel<<<(1024 * 512 + 255) / 256, 256, 0, stream>>>(fc1w, wf1_t, 512, 1024);
  wcvt_kernel<<<(1024 * 512 + 255) / 256, 256, 0, stream>>>(fc2w, wf2_t, 1024, 512);

  // LN1 + roll + window partition
  ln1_win_kernel<<<NTOK / 8, 256, 0, stream>>>(x, ln1w, ln1b, winX);

  // QKV: M=32768, N=1536, K=512
  gemm_qkv<<<dim3(1536 / 128, 32768 / 128), 256, 0, stream>>>(winX, wq_t, qkvb, qb, kb, vb);

  // attention: 512 windows * 16 heads
  attn_kernel<<<NWIN * 16, 128, 0, stream>>>(qb, kb, vb, attnO);

  // proj: M=32768, N=512, K=512 (+ window reverse + residual)
  gemm_proj<<<dim3(512 / 128, 32768 / 128), 256, 0, stream>>>(attnO, wp_t, projb, x, x2);

  // LN2
  ln2_kernel<<<NTOK / 8, 256, 0, stream>>>(x2, ln2w, ln2b, z16);

  // FC1: M=25088, N=1024, K=512 (+GELU)
  gemm_fc1<<<dim3(1024 / 128, 25088 / 128), 256, 0, stream>>>(z16, wf1_t, fc1b, h16);

  // FC2: M=25088, N=512, K=1024 (+bias+residual)
  gemm_fc2<<<dim3(512 / 128, 25088 / 128), 256, 0, stream>>>(h16, wf2_t, fc2b, x2, out);
}
